// TC_base_25262997635754
// MI455X (gfx1250) — matrix operand order repaired and checked by reading, not compiled, not run on hardware
//
#include <hip/hip_runtime.h>
#include <hip/hip_bf16.h>

typedef __attribute__((ext_vector_type(16))) _Float16 v16h;
typedef __attribute__((ext_vector_type(8)))  float    v8f;

#define B_   16
#define S_   512
#define D_   768
#define H_   384
#define G4_  1536   // 4*H
#define E_   768
#define NCLS_ 2
#define EPS_ 1e-5f

__device__ __forceinline__ float sigm(float x){ return 1.0f/(1.0f+__expf(-x)); }

// ---------------- WMMA fragment loaders (wave32, 16x16x32 f16) ----------------
// A (16xK slice, row-major, f32 source -> f16): lane l: m=l&15, kh=l>>4
//   elements 0..7  = A[m][k0 + kh*8 + 0..7]
//   elements 8..15 = A[m][k0 + 16 + kh*8 + 0..7]
__device__ __forceinline__ v16h load_a_f32(const float* base, int lda){
  const int lane = threadIdx.x & 31;
  const int m = lane & 15, kh = lane >> 4;
  const float* p0 = base + (size_t)m*lda + kh*8;
  const float* p1 = p0 + 16;
  v16h a;
  #pragma unroll
  for (int i=0;i<8;++i) a[i]   = (_Float16)p0[i];
  #pragma unroll
  for (int i=0;i<8;++i) a[8+i] = (_Float16)p1[i];
  return a;
}
__device__ __forceinline__ v16h load_a_f16(const _Float16* base, int lda){
  const int lane = threadIdx.x & 31;
  const int m = lane & 15, kh = lane >> 4;
  const _Float16* p0 = base + (size_t)m*lda + kh*8;
  const _Float16* p1 = p0 + 16;
  v16h a;
  #pragma unroll
  for (int i=0;i<8;++i) a[i]   = p0[i];
  #pragma unroll
  for (int i=0;i<8;++i) a[8+i] = p1[i];
  return a;
}
// B (Kx16 slice from row-major W[n][k]): lane l: n=l&15, kh=l>>4
//   elements 0..7  = W[n][k0 + kh*8 + 0..7]
//   elements 8..15 = W[n][k0 + 16 + kh*8 + 0..7]
__device__ __forceinline__ v16h load_b_f16(const _Float16* base, int ldk){
  const int lane = threadIdx.x & 31;
  const int n = lane & 15, kh = lane >> 4;
  const _Float16* p = base + (size_t)n*ldk + kh*8;
  v16h b;
  #pragma unroll
  for (int i=0;i<8;++i) b[i] = p[i];
  #pragma unroll
  for (int i=0;i<8;++i) b[8+i] = p[16+i];
  return b;
}

// ---------------- weight f32 -> f16 ----------------
__global__ __launch_bounds__(256) void cvt_f16(const float* __restrict__ src,
                                               _Float16* __restrict__ dst, int n){
  int i = blockIdx.x*256 + threadIdx.x;
  if (i < n) dst[i] = (_Float16)src[i];
}

// ---------------- C[M,N] = A(f32)[M,K] @ W16[N,K]^T + bias ----------------
// one wave per block; wave computes a 16x64 tile
__global__ __launch_bounds__(32) void gemm_aWt(
    const float* __restrict__ A, const _Float16* __restrict__ W,
    const float* __restrict__ bias, float* __restrict__ C, int K, int N)
{
  const int m0 = blockIdx.x * 16;
  const int n0 = blockIdx.y * 64;
  const int lane = threadIdx.x & 31;
  const float* Abase = A + (size_t)m0 * K;
  const _Float16* W0 = W + (size_t)(n0+ 0)*K;
  const _Float16* W1 = W + (size_t)(n0+16)*K;
  const _Float16* W2 = W + (size_t)(n0+32)*K;
  const _Float16* W3 = W + (size_t)(n0+48)*K;
  v8f acc0={},acc1={},acc2={},acc3={};
  for (int k0 = 0; k0 < K; k0 += 32) {
    __builtin_prefetch(W0 + k0 + 256, 0, 0);
    v16h a  = load_a_f32(Abase + k0, K);
    v16h b0 = load_b_f16(W0 + k0, K);
    v16h b1 = load_b_f16(W1 + k0, K);
    v16h b2 = load_b_f16(W2 + k0, K);
    v16h b3 = load_b_f16(W3 + k0, K);
    acc0 = __builtin_amdgcn_wmma_f32_16x16x32_f16(false,a,false,b0,(short)0,acc0,false,false);
    acc1 = __builtin_amdgcn_wmma_f32_16x16x32_f16(false,a,false,b1,(short)0,acc1,false,false);
    acc2 = __builtin_amdgcn_wmma_f32_16x16x32_f16(false,a,false,b2,(short)0,acc2,false,false);
    acc3 = __builtin_amdgcn_wmma_f32_16x16x32_f16(false,a,false,b3,(short)0,acc3,false,false);
  }
  const int n = lane & 15, mh = (lane >> 4) << 3;
  const float bv0 = bias[n0+n], bv1 = bias[n0+16+n], bv2 = bias[n0+32+n], bv3 = bias[n0+48+n];
  #pragma unroll
  for (int r=0;r<8;++r){
    float* crow = C + (size_t)(m0 + mh + r)*N + n0 + n;
    crow[0]  = acc0[r] + bv0;
    crow[16] = acc1[r] + bv1;
    crow[32] = acc2[r] + bv2;
    crow[48] = acc3[r] + bv3;
  }
}

// ---------------- batched C[b][M,N] = A16[b][M,K] @ B16[b][N,K]^T ----------------
__global__ __launch_bounds__(32) void gemm_ab16(
    const _Float16* __restrict__ A, const _Float16* __restrict__ Bm,
    float* __restrict__ C, int K, int N, int Ma)
{
  const int b  = blockIdx.z;
  const int m0 = blockIdx.x * 16;
  const int n0 = blockIdx.y * 64;
  const int lane = threadIdx.x & 31;
  const _Float16* Ab = A  + (size_t)b*Ma*K + (size_t)m0*K;
  const _Float16* B0 = Bm + (size_t)b*N*K + (size_t)(n0+ 0)*K;
  const _Float16* B1 = Bm + (size_t)b*N*K + (size_t)(n0+16)*K;
  const _Float16* B2 = Bm + (size_t)b*N*K + (size_t)(n0+32)*K;
  const _Float16* B3 = Bm + (size_t)b*N*K + (size_t)(n0+48)*K;
  v8f acc0={},acc1={},acc2={},acc3={};
  for (int k0 = 0; k0 < K; k0 += 32) {
    v16h a  = load_a_f16(Ab + k0, K);
    v16h b0 = load_b_f16(B0 + k0, K);
    v16h b1 = load_b_f16(B1 + k0, K);
    v16h b2 = load_b_f16(B2 + k0, K);
    v16h b3 = load_b_f16(B3 + k0, K);
    acc0 = __builtin_amdgcn_wmma_f32_16x16x32_f16(false,a,false,b0,(short)0,acc0,false,false);
    acc1 = __builtin_amdgcn_wmma_f32_16x16x32_f16(false,a,false,b1,(short)0,acc1,false,false);
    acc2 = __builtin_amdgcn_wmma_f32_16x16x32_f16(false,a,false,b2,(short)0,acc2,false,false);
    acc3 = __builtin_amdgcn_wmma_f32_16x16x32_f16(false,a,false,b3,(short)0,acc3,false,false);
  }
  const int n = lane & 15, mh = (lane >> 4) << 3;
  #pragma unroll
  for (int r=0;r<8;++r){
    float* crow = C + ((size_t)b*Ma + m0 + mh + r)*N + n0 + n;
    crow[0]=acc0[r]; crow[16]=acc1[r]; crow[32]=acc2[r]; crow[48]=acc3[r];
  }
}

// ---------------- in-place row LayerNorm ----------------
__global__ __launch_bounds__(256) void ln_rows(
    float* __restrict__ X, const float* __restrict__ g,
    const float* __restrict__ bta, int N)
{
  float* x = X + (size_t)blockIdx.x * N;
  const int tid = threadIdx.x;
  float s=0.f, ss=0.f;
  for (int i=tid; i<N; i+=256){ float v=x[i]; s+=v; ss+=v*v; }
  __shared__ float r1[256], r2[256];
  r1[tid]=s; r2[tid]=ss; __syncthreads();
  for (int st=128; st>0; st>>=1){
    if (tid<st){ r1[tid]+=r1[tid+st]; r2[tid]+=r2[tid+st]; }
    __syncthreads();
  }
  const float mean = r1[0] / (float)N;
  const float var  = r2[0] / (float)N - mean*mean;
  const float rstd = rsqrtf(var + EPS_);
  for (int i=tid; i<N; i+=256){ float v=x[i]; x[i] = (v-mean)*rstd*g[i] + bta[i]; }
}

// ---------------- BiLSTM scan: one persistent workgroup per direction ----------------
// 24 waves; wave w owns columns j=w*16..w*16+15 across all four gates -> c stays in regs
__global__ __launch_bounds__(768) void lstm_scan(
    const float* __restrict__ xg, const _Float16* __restrict__ Wh16,
    const float* __restrict__ bh,
    const float* __restrict__ lnh_g, const float* __restrict__ lnh_b,
    const float* __restrict__ lnc_g, const float* __restrict__ lnc_b,
    float* __restrict__ out)
{
  const int dir  = blockIdx.x;          // 0 = forward, 1 = backward
  const int wave = threadIdx.x >> 5;
  const int lane = threadIdx.x & 31;
  const int n    = lane & 15;
  const int mh   = (lane >> 4) << 3;    // 0 or 8 (row offset)

  __shared__ _Float16 h_sh[16][H_ + 16];          // padded lda = 400
  __shared__ float zsum[24][16], zss[24][16];
  __shared__ float csum[24][16], css[24][16];
  __shared__ float stZ[16][2], stC[16][2];

  for (int i = threadIdx.x; i < 16*(H_+16); i += blockDim.x)
    (&h_sh[0][0])[i] = (_Float16)0.f;

  const int j     = wave*16 + n;                  // column within H
  const int gcol0 = j;                            // i-gate
  const int gcol1 = H_   + j;                     // f
  const int gcol2 = 2*H_ + j;                     // g
  const int gcol3 = 3*H_ + j;                     // o
  const float bh0=bh[gcol0], bh1=bh[gcol1], bh2=bh[gcol2], bh3=bh[gcol3];
  const float hg0=lnh_g[gcol0], hg1=lnh_g[gcol1], hg2=lnh_g[gcol2], hg3=lnh_g[gcol3];
  const float hb0=lnh_b[gcol0], hb1=lnh_b[gcol1], hb2=lnh_b[gcol2], hb3=lnh_b[gcol3];
  const float cgv=lnc_g[j], cbv=lnc_b[j];
  const _Float16* W0 = Wh16 + (size_t)(wave*16       )*H_;
  const _Float16* W1 = Wh16 + (size_t)(H_   + wave*16)*H_;
  const _Float16* W2 = Wh16 + (size_t)(2*H_ + wave*16)*H_;
  const _Float16* W3 = Wh16 + (size_t)(3*H_ + wave*16)*H_;

  v8f c = {};
  for (int step = 0; step < S_; ++step) {
    const int tt = dir ? (S_-1-step) : step;
    __syncthreads();                               // h_sh ready / stat arrays free

    // z = h @ Wh^T for this wave's 4 gate tiles
    v8f z0={},z1={},z2={},z3={};
    for (int k0 = 0; k0 < H_; k0 += 32) {
      v16h a  = load_a_f16(&h_sh[0][0] + k0, H_+16);
      v16h b0 = load_b_f16(W0 + k0, H_);
      v16h b1 = load_b_f16(W1 + k0, H_);
      v16h b2 = load_b_f16(W2 + k0, H_);
      v16h b3 = load_b_f16(W3 + k0, H_);
      z0 = __builtin_amdgcn_wmma_f32_16x16x32_f16(false,a,false,b0,(short)0,z0,false,false);
      z1 = __builtin_amdgcn_wmma_f32_16x16x32_f16(false,a,false,b1,(short)0,z1,false,false);
      z2 = __builtin_amdgcn_wmma_f32_16x16x32_f16(false,a,false,b2,(short)0,z2,false,false);
      z3 = __builtin_amdgcn_wmma_f32_16x16x32_f16(false,a,false,b3,(short)0,z3,false,false);
    }
    #pragma unroll
    for (int r=0;r<8;++r){ z0[r]+=bh0; z1[r]+=bh1; z2[r]+=bh2; z3[r]+=bh3; }

    // per-row partial stats over the 1536 gate dim
    #pragma unroll
    for (int r=0;r<8;++r){
      float p  = z0[r]+z1[r]+z2[r]+z3[r];
      float ps = z0[r]*z0[r]+z1[r]*z1[r]+z2[r]*z2[r]+z3[r]*z3[r];
      p  += __shfl_xor(p ,1,32); p  += __shfl_xor(p ,2,32); p  += __shfl_xor(p ,4,32); p  += __shfl_xor(p ,8,32);
      ps += __shfl_xor(ps,1,32); ps += __shfl_xor(ps,2,32); ps += __shfl_xor(ps,4,32); ps += __shfl_xor(ps,8,32);
      if (n == 0){ zsum[wave][mh+r]=p; zss[wave][mh+r]=ps; }
    }
    __syncthreads();
    if (threadIdx.x < 16){
      float s=0.f, ss=0.f;
      for (int w=0; w<24; ++w){ s += zsum[w][threadIdx.x]; ss += zss[w][threadIdx.x]; }
      float mean = s * (1.0f/(float)G4_);
      float var  = ss * (1.0f/(float)G4_) - mean*mean;
      stZ[threadIdx.x][0]=mean; stZ[threadIdx.x][1]=rsqrtf(var+EPS_);
    }
    __syncthreads();

    // gates = x_t + LN(z); cell update (c stays in registers)
    float o_v[8];
    #pragma unroll
    for (int r=0;r<8;++r){
      const int m = mh + r;
      const float mean = stZ[m][0], rstd = stZ[m][1];
      const float* xrow = xg + ((size_t)m*S_ + tt)*G4_;
      float gi = xrow[gcol0] + (z0[r]-mean)*rstd*hg0 + hb0;
      float gf = xrow[gcol1] + (z1[r]-mean)*rstd*hg1 + hb1;
      float gg = xrow[gcol2] + (z2[r]-mean)*rstd*hg2 + hb2;
      float go = xrow[gcol3] + (z3[r]-mean)*rstd*hg3 + hb3;
      c[r] = sigm(gf)*c[r] + sigm(gi)*tanhf(gg);
      o_v[r] = sigm(go);
    }

    // per-row partial stats of c over H
    #pragma unroll
    for (int r=0;r<8;++r){
      float p = c[r], ps = c[r]*c[r];
      p  += __shfl_xor(p ,1,32); p  += __shfl_xor(p ,2,32); p  += __shfl_xor(p ,4,32); p  += __shfl_xor(p ,8,32);
      ps += __shfl_xor(ps,1,32); ps += __shfl_xor(ps,2,32); ps += __shfl_xor(ps,4,32); ps += __shfl_xor(ps,8,32);
      if (n == 0){ csum[wave][mh+r]=p; css[wave][mh+r]=ps; }
    }
    __syncthreads();
    if (threadIdx.x < 16){
      float s=0.f, ss=0.f;
      for (int w=0; w<24; ++w){ s += csum[w][threadIdx.x]; ss += css[w][threadIdx.x]; }
      float mean = s * (1.0f/(float)H_);
      float var  = ss * (1.0f/(float)H_) - mean*mean;
      stC[threadIdx.x][0]=mean; stC[threadIdx.x][1]=rsqrtf(var+EPS_);
    }
    __syncthreads();

    // h = sig(o) * tanh(LN(c)); publish to LDS + global
    #pragma unroll
    for (int r=0;r<8;++r){
      const int m = mh + r;
      float cn = (c[r]-stC[m][0])*stC[m][1]*cgv + cbv;
      float hv = o_v[r]*tanhf(cn);
      h_sh[m][j] = (_Float16)hv;
      out[((size_t)m*S_ + tt)*E_ + dir*H_ + j] = hv;
    }
  }
}

// ---------------- attention scalars: kw = kx@w1, qw = qx@w2 ----------------
__global__ __launch_bounds__(32) void attn_vec(
    const float* __restrict__ kx, const float* __restrict__ qx,
    const float* __restrict__ attn_w, float* __restrict__ kw, float* __restrict__ qw)
{
  const int row = blockIdx.x;
  const int lane = threadIdx.x;
  const float* w1 = attn_w;
  const float* w2 = attn_w + E_;
  const float* kr = kx + (size_t)row * E_;
  const float* qr = qx + (size_t)row * E_;
  float sk=0.f, sq=0.f;
  for (int i=lane; i<E_; i+=32){ sk += kr[i]*w1[i]; sq += qr[i]*w2[i]; }
  #pragma unroll
  for (int m=16;m>=1;m>>=1){ sk += __shfl_xor(sk,m,32); sq += __shfl_xor(sq,m,32); }
  if (lane==0){ kw[row]=sk; qw[row]=sq; }
}

// ---------------- P[b][q][k] = softmax_k tanh(qw[b,q] + kw[b,k]), f16 ----------------
__global__ __launch_bounds__(256) void score_softmax(
    const float* __restrict__ kw, const float* __restrict__ qw, _Float16* __restrict__ P)
{
  const int bq  = blockIdx.x;                 // b*512 + q
  const int b   = bq >> 9;
  const int tid = threadIdx.x;
  const float qv = qw[bq];
  const float* kwb = kw + (size_t)b * S_;
  float t0 = tanhf(qv + kwb[tid]);
  float t1 = tanhf(qv + kwb[tid + 256]);
  __shared__ float red[256];
  red[tid] = fmaxf(t0, t1); __syncthreads();
  for (int s=128; s>0; s>>=1){ if (tid<s) red[tid] = fmaxf(red[tid], red[tid+s]); __syncthreads(); }
  const float mx = red[0]; __syncthreads();
  float e0 = __expf(t0-mx), e1 = __expf(t1-mx);
  red[tid] = e0+e1; __syncthreads();
  for (int s=128; s>0; s>>=1){ if (tid<s) red[tid] += red[tid+s]; __syncthreads(); }
  const float inv = 1.0f / red[0];
  _Float16* prow = P + (size_t)bq * S_;
  prow[tid]     = (_Float16)(e0*inv);
  prow[tid+256] = (_Float16)(e1*inv);
}

// ---------------- kxT[b][e][s] = (f16) kx[b][s][e] ----------------
__global__ __launch_bounds__(256) void transpose_to_f16(
    const float* __restrict__ kx, _Float16* __restrict__ kxT)
{
  int idx = blockIdx.x*256 + threadIdx.x;
  if (idx >= B_*E_*S_) return;
  int s = idx & (S_-1);
  int e = (idx >> 9) % E_;
  int b = idx / (E_*S_);
  kxT[idx] = (_Float16)kx[((size_t)b*S_ + s)*E_ + e];
}

// ---------------- scc_mean[b][e] = sum_t scc[b][t][e] / len[b] ----------------
__global__ __launch_bounds__(256) void scc_mean_k(
    const float* __restrict__ scc, const int* __restrict__ len, float* __restrict__ sm)
{
  int idx = blockIdx.x*256 + threadIdx.x;
  if (idx >= B_*E_) return;
  int b = idx / E_;
  const float* p = scc + (size_t)b*S_*E_ + (idx % E_);
  float s = 0.f;
  for (int t=0; t<S_; ++t) s += p[(size_t)t*E_];
  sm[idx] = s / (float)len[b];
}

// ---------------- out[b][c] = sm[b]·Wc[c] + bc[c] ----------------
__global__ __launch_bounds__(32) void final_fc(
    const float* __restrict__ sm, const float* __restrict__ Wc,
    const float* __restrict__ bc, float* __restrict__ out)
{
  const int o = blockIdx.x;   // b*NCLS + c
  const int b = o / NCLS_, cc = o % NCLS_;
  const int lane = threadIdx.x;
  const float* x = sm + (size_t)b*E_;
  const float* w = Wc + (size_t)cc*E_;
  float s=0.f;
  for (int i=lane;i<E_;i+=32) s += x[i]*w[i];
  #pragma unroll
  for (int m=16;m>=1;m>>=1) s += __shfl_xor(s,m,32);
  if (lane==0) out[o] = s + bc[cc];
}

// ======================================================================
extern "C" void kernel_launch(void* const* d_in, const int* in_sizes, int n_in,
                              void* d_out, int out_size, void* d_ws, size_t ws_size,
                              hipStream_t stream)
{
  (void)in_sizes; (void)n_in; (void)out_size; (void)ws_size;
  const float* features = (const float*)d_in[0];
  const int*   ids_len  = (const int*)  d_in[1];
  const float* Wx   = (const float*)d_in[2];
  const float* bx   = (const float*)d_in[3];
  const float* Wh   = (const float*)d_in[4];
  const float* bh   = (const float*)d_in[5];
  const float* lnx_g= (const float*)d_in[6];
  const float* lnx_b= (const float*)d_in[7];
  const float* lnh_g= (const float*)d_in[8];
  const float* lnh_b= (const float*)d_in[9];
  const float* lnc_g= (const float*)d_in[10];
  const float* lnc_b= (const float*)d_in[11];
  const float* Wk   = (const float*)d_in[12];
  const float* bk   = (const float*)d_in[13];
  const float* Wq   = (const float*)d_in[14];
  const float* bq   = (const float*)d_in[15];
  const float* Wp   = (const float*)d_in[16];
  const float* bp   = (const float*)d_in[17];
  const float* attn_w = (const float*)d_in[18];
  const float* Wc   = (const float*)d_in[19];
  const float* bc   = (const float*)d_in[20];
  float* out = (float*)d_out;

  char* ws = (char*)d_ws;
  size_t off = 0;
  auto take = [&](size_t bytes)->char*{
    char* p = ws + off; off += (bytes + 255) & ~(size_t)255; return p;
  };
  _Float16* Wx16 = (_Float16*)take((size_t)G4_*D_*2);
  _Float16* Wh16 = (_Float16*)take((size_t)G4_*H_*2);
  _Float16* Wk16 = (_Float16*)take((size_t)E_*E_*2);
  _Float16* Wq16 = (_Float16*)take((size_t)E_*E_*2);
  _Float16* Wp16 = (_Float16*)take((size_t)E_*E_*2);
  float*    xg   = (float*)   take((size_t)B_*S_*G4_*4);  // reused as scc later
  float*    lout = (float*)   take((size_t)B_*S_*E_*4);
  float*    kx   = (float*)   take((size_t)B_*S_*E_*4);
  float*    qx   = (float*)   take((size_t)B_*S_*E_*4);
  float*    attn = (float*)   take((size_t)B_*S_*E_*4);
  _Float16* P    = (_Float16*)take((size_t)B_*S_*S_*2);
  _Float16* kxT  = (_Float16*)take((size_t)B_*E_*S_*2);
  float*    kw   = (float*)   take((size_t)B_*S_*4);
  float*    qw   = (float*)   take((size_t)B_*S_*4);
  float*    sm   = (float*)   take((size_t)B_*E_*4);
  float*    scc  = xg;  // xg dead after lstm_scan

  // 1) convert weights to f16 once
  cvt_f16<<<(G4_*D_+255)/256,256,0,stream>>>(Wx, Wx16, G4_*D_);
  cvt_f16<<<(G4_*H_+255)/256,256,0,stream>>>(Wh, Wh16, G4_*H_);
  cvt_f16<<<(E_*E_+255)/256,256,0,stream>>>(Wk, Wk16, E_*E_);
  cvt_f16<<<(E_*E_+255)/256,256,0,stream>>>(Wq, Wq16, E_*E_);
  cvt_f16<<<(E_*E_+255)/256,256,0,stream>>>(Wp, Wp16, E_*E_);

  // 2) xg = LN(features @ Wx^T + bx)
  gemm_aWt<<<dim3(B_*S_/16, G4_/64),32,0,stream>>>(features, Wx16, bx, xg, D_, G4_);
  ln_rows<<<B_*S_,256,0,stream>>>(xg, lnx_g, lnx_b, G4_);

  // 3) bidirectional LSTM scan (grid.x = direction)
  lstm_scan<<<2,768,0,stream>>>(xg, Wh16, bh, lnh_g, lnh_b, lnc_g, lnc_b, lout);

  // 4) kx, qx projections
  gemm_aWt<<<dim3(B_*S_/16, E_/64),32,0,stream>>>(lout, Wk16, bk, kx, E_, E_);
  gemm_aWt<<<dim3(B_*S_/16, E_/64),32,0,stream>>>(lout, Wq16, bq, qx, E_, E_);

  // 5) attention
  attn_vec<<<B_*S_,32,0,stream>>>(kx, qx, attn_w, kw, qw);
  score_softmax<<<B_*S_,256,0,stream>>>(kw, qw, P);
  transpose_to_f16<<<(B_*E_*S_+255)/256,256,0,stream>>>(kx, kxT);
  gemm_ab16<<<dim3(S_/16, E_/64, B_),32,0,stream>>>(P, kxT, attn, S_, E_, S_);

  // 6) scc = attn @ Wp^T + bp  (into xg alias), mean-pool, classifier
  gemm_aWt<<<dim3(B_*S_/16, E_/64),32,0,stream>>>(attn, Wp16, bp, scc, E_, E_);
  scc_mean_k<<<(B_*E_+255)/256,256,0,stream>>>(scc, ids_len, sm);
  final_fc<<<B_*NCLS_,32,0,stream>>>(sm, Wc, bc, out);
}
